// FeatureAttention_20830591386239
// MI455X (gfx1250) — compile-verified, matrix operand order checked by reading, not run on hardware
//
#include <hip/hip_runtime.h>
#include <hip/hip_bf16.h>
#include <math.h>

// ---------------- CDNA5 WMMA types / fragment maps ----------------
typedef __attribute__((ext_vector_type(16))) _Float16 v16h;
typedef __attribute__((ext_vector_type(8)))  float    v8f;

union Frag16 { v16h h; unsigned u[8]; _Float16 e[16]; };

// 16-bit A/B matrix: lane l holds row/col (l%16); half j covers
// K = j + (j>=8 ? 8 : 0) + (l>=16 ? 8 : 0).  Pairs (2jj,2jj+1) are K-contiguous.
__device__ __forceinline__ int kpair(int jj, int hi) {
  return 2 * jj + ((jj >= 4) ? 8 : 0) + (hi ? 8 : 0);
}
__device__ __forceinline__ int kone(int j, int hi) {
  return j + ((j >= 8) ? 8 : 0) + (hi ? 8 : 0);
}

#define NTOK  1466   // tokens per scene = 2*(4+729)
#define MROWS 2932   // S*NTOK
#define MPAD  2944   // 23*128
#define NKV   1472   // 92*16 / 46*32 padded kv length
#define EMB   1024
#define NHEAD 16
#define HDIM  64

// ---------------- token assembly ----------------
__global__ void build_tokens_kernel(const float* __restrict__ pt,
                                    const float* __restrict__ reg,
                                    float* __restrict__ X) {
  int idx = blockIdx.x * blockDim.x + threadIdx.x;
  if (idx >= MROWS * EMB) return;
  int m = idx >> 10, c = idx & 1023;
  int s = m / NTOK, n = m % NTOK;
  int l = n / 733, r = n % 733;
  float v;
  if (r < 4) {
    int slot = (l == 0) ? 0 : 1;
    v = reg[(slot * 4 + r) * EMB + c];
  } else {
    v = pt[(((size_t)(l * 2 + s)) * 729 + (r - 4)) * EMB + c];
  }
  X[idx] = v;
}

// ---------------- weight convert + transpose: (K x N) f32 -> (N x K) f16 -----
__global__ void convt_kernel(const float* __restrict__ in, _Float16* __restrict__ out,
                             int K, int N) {
  int idx = blockIdx.x * blockDim.x + threadIdx.x;
  if (idx >= K * N) return;
  int k = idx / N, n = idx % N;
  out[(size_t)n * K + k] = (_Float16)in[idx];
}

// ---------------- layernorm over C=1024, output f16 into A16 (lda=4096) -----
__global__ __launch_bounds__(256) void layernorm_f16_kernel(
    const float* __restrict__ X, const float* __restrict__ w,
    const float* __restrict__ b, _Float16* __restrict__ out) {
  int m = blockIdx.x;
  int t = threadIdx.x;
  const float* x = X + (size_t)m * EMB;
  float v[4], s = 0.f, s2 = 0.f;
  for (int i = 0; i < 4; i++) {
    v[i] = x[t + i * 256];
    s += v[i]; s2 += v[i] * v[i];
  }
  for (int o = 1; o < 32; o <<= 1) { s += __shfl_xor(s, o, 32); s2 += __shfl_xor(s2, o, 32); }
  __shared__ float ss[8], ss2[8];
  if ((t & 31) == 0) { ss[t >> 5] = s; ss2[t >> 5] = s2; }
  __syncthreads();
  float ts = 0.f, ts2 = 0.f;
  for (int i = 0; i < 8; i++) { ts += ss[i]; ts2 += ss2[i]; }
  float mean = ts * (1.f / 1024.f);
  float var  = ts2 * (1.f / 1024.f) - mean * mean;
  float inv  = rsqrtf(var + 1e-5f);
  _Float16* o16 = out + (size_t)m * 4096;
  for (int i = 0; i < 4; i++) {
    int c = t + i * 256;
    o16[c] = (_Float16)((v[i] - mean) * inv * w[c] + b[c]);
  }
}

// ---------------- WMMA GEMM: C[MPAD x N] = A[MPAD x K](f16,lda=4096) * W^T + bias
// Bt is N x K (row-major), i.e. the transposed weight; 128x128x32 tiles.
// Tiles are staged with CDNA5 GLOBAL_LOAD_ASYNC_TO_LDS_B128 (ASYNCcnt) and
// double-buffered so the next tile streams in while WMMAs consume the current.
__global__ __launch_bounds__(256) void gemm_f16_wmma_kernel(
    const _Float16* __restrict__ A, const _Float16* __restrict__ Bt,
    float* __restrict__ C, const float* __restrict__ bias, int N, int K) {
  __shared__ __align__(16) _Float16 As[2][128 * 32];
  __shared__ __align__(16) _Float16 Bs[2][128 * 32];   // Bs[n_local*32 + k]
  const int tid  = threadIdx.x;
  const int lane = tid & 31, wid = tid >> 5;
  const int wm = wid >> 1, wn = wid & 1;
  const int hi = (lane >= 16) ? 1 : 0;
  const int l16 = lane & 15;
  const int bm = blockIdx.y * 128, bn = blockIdx.x * 128;

  v8f acc[2][4];
  for (int i = 0; i < 2; i++)
    for (int j = 0; j < 4; j++)
      for (int q = 0; q < 8; q++) acc[i][j][q] = 0.f;

  const int rA  = tid >> 1, segA = (tid & 1) * 16;
  const int cB  = tid >> 1, segB = (tid & 1) * 16;

  // Issue one 32B A segment + one 32B B segment per thread into LDS buffer buf.
  // offset:16 advances both the LDS and memory addresses (ISA 08_async_tensor §4.4).
  auto issue_tile = [&](int buf, int kk) {
    unsigned la = (unsigned)(uintptr_t)&As[buf][rA * 32 + segA];
    unsigned lb = (unsigned)(uintptr_t)&Bs[buf][cB * 32 + segB];
    unsigned ga = ((unsigned)(bm + rA) * 4096u + (unsigned)(kk + segA)) * 2u;
    unsigned gb = ((unsigned)(bn + cB) * (unsigned)K + (unsigned)(kk + segB)) * 2u;
    asm volatile("global_load_async_to_lds_b128 %0, %1, %2\n\t"
                 "global_load_async_to_lds_b128 %0, %1, %2 offset:16"
                 :: "v"(la), "v"(ga), "s"(A) : "memory");
    asm volatile("global_load_async_to_lds_b128 %0, %1, %2\n\t"
                 "global_load_async_to_lds_b128 %0, %1, %2 offset:16"
                 :: "v"(lb), "v"(gb), "s"(Bt) : "memory");
  };

  const int nsteps = K / 32;
  issue_tile(0, 0);
  for (int st = 0; st < nsteps; st++) {
    const int buf = st & 1;
    if (st + 1 < nsteps) {
      issue_tile(buf ^ 1, (st + 1) * 32);
      asm volatile("s_wait_asynccnt 0x4" ::: "memory");  // current tile's 4 done
    } else {
      asm volatile("s_wait_asynccnt 0x0" ::: "memory");
    }
    __syncthreads();

    Frag16 af[2];
    for (int mt = 0; mt < 2; mt++) {
      const _Float16* p = &As[buf][(wm * 32 + mt * 16 + l16) * 32];
      for (int jj = 0; jj < 8; jj++) af[mt].u[jj] = *(const unsigned*)&p[kpair(jj, hi)];
    }
    for (int nt = 0; nt < 4; nt++) {
      Frag16 bf;
      const _Float16* p = &Bs[buf][(wn * 64 + nt * 16 + l16) * 32];
      for (int jj = 0; jj < 8; jj++) bf.u[jj] = *(const unsigned*)&p[kpair(jj, hi)];
      for (int mt = 0; mt < 2; mt++)
        acc[mt][nt] = __builtin_amdgcn_wmma_f32_16x16x32_f16(
            false, af[mt].h, false, bf.h, (short)0, acc[mt][nt], false, false);
    }
    __syncthreads();
  }

  for (int mt = 0; mt < 2; mt++)
    for (int nt = 0; nt < 4; nt++) {
      int col = bn + wn * 64 + nt * 16 + l16;
      float bv = bias ? bias[col] : 0.f;
      for (int j = 0; j < 8; j++) {
        int row = bm + wm * 32 + mt * 16 + j + hi * 8;
        C[(size_t)row * N + col] = acc[mt][nt][j] + bv;
      }
    }
}

// ---------------- qk-norm + 2D RoPE + pack q/k/v into (s*H+h, NKV, 64) f16 ----
__global__ __launch_bounds__(32) void qkv_pack_rope_kernel(
    const float* __restrict__ QKV,   // MPAD x 3072, rows<MROWS valid
    const float* __restrict__ qw, const float* __restrict__ qb,
    const float* __restrict__ kw, const float* __restrict__ kb,
    _Float16* __restrict__ q16, _Float16* __restrict__ k16,
    _Float16* __restrict__ v16) {
  int m = blockIdx.x, h = blockIdx.y, t = threadIdx.x;
  int s = m / NTOK, n = m % NTOK;
  const float* base = QKV + (size_t)m * 3072 + h * HDIM;
  float q0 = base[t],        q1 = base[t + 32];
  float k0 = base[1024 + t], k1 = base[1024 + t + 32];
  float v0 = base[2048 + t], v1 = base[2048 + t + 32];

  // per-head layernorm over 64 dims (two elems per lane, wave32 reduce)
  auto hln = [&](float a, float b2, float& oa, float& ob,
                 const float* w, const float* bb) {
    float s1 = a + b2, sq = a * a + b2 * b2;
    for (int o = 1; o < 32; o <<= 1) { s1 += __shfl_xor(s1, o, 32); sq += __shfl_xor(sq, o, 32); }
    float mean = s1 * (1.f / 64.f), var = sq * (1.f / 64.f) - mean * mean;
    float inv = rsqrtf(var + 1e-5f);
    oa = (a  - mean) * inv * w[t]      + bb[t];
    ob = (b2 - mean) * inv * w[t + 32] + bb[t + 32];
  };
  float qa, qb2, ka, kb2;
  hln(q0, q1, qa, qb2, qw, qb);
  hln(k0, k1, ka, kb2, kw, kb);

  // 2D RoPE: dims 0..31 use y-pos, 32..63 use x-pos; registers at pos (0,0)
  int l = n / 733, r = n % 733;
  float py = 0.f, px = 0.f;
  if (r >= 4) { int pi = r - 4; py = (float)(pi / 27 + 1); px = (float)(pi % 27 + 1); }
  int fi = t & 15;
  float freq = powf(100.f, -((float)fi) / 16.f);
  float c0 = cosf(py * freq), s0 = sinf(py * freq);
  float c1 = cosf(px * freq), s1v = sinf(px * freq);
  float qp0 = __shfl_xor(qa, 16, 32), qp1 = __shfl_xor(qb2, 16, 32);
  float kp0 = __shfl_xor(ka, 16, 32), kp1 = __shfl_xor(kb2, 16, 32);
  float sgn = (t < 16) ? -1.f : 1.f;
  float qr0 = qa  * c0 + sgn * qp0 * s0;
  float qr1 = qb2 * c1 + sgn * qp1 * s1v;
  float kr0 = ka  * c0 + sgn * kp0 * s0;
  float kr1 = kb2 * c1 + sgn * kp1 * s1v;

  size_t ob = ((size_t)(s * NHEAD + h) * NKV + n) * HDIM;
  q16[ob + t] = (_Float16)qr0;  q16[ob + t + 32] = (_Float16)qr1;
  k16[ob + t] = (_Float16)kr0;  k16[ob + t + 32] = (_Float16)kr1;
  v16[ob + t] = (_Float16)v0;   v16[ob + t + 32] = (_Float16)v1;
}

// ---------------- flash attention (one wave per scene*head*16-row q tile) ----
__global__ __launch_bounds__(32) void flash_attn_kernel(
    const _Float16* __restrict__ q16, const _Float16* __restrict__ k16,
    const _Float16* __restrict__ v16, _Float16* __restrict__ outA) {
  int qt = blockIdx.x;                 // 0..91
  int sh = blockIdx.y;                 // s*16+h
  int lane = threadIdx.x;
  int hi = (lane >= 16) ? 1 : 0, l16 = lane & 15;
  size_t base = (size_t)sh * NKV * HDIM;

  Frag16 qa[2];
  {
    const _Float16* qp = q16 + base + (size_t)(qt * 16 + l16) * HDIM;
    for (int part = 0; part < 2; part++)
      for (int jj = 0; jj < 8; jj++)
        qa[part].u[jj] = *(const unsigned*)&qp[kpair(jj, hi) + part * 32];
  }

  float mrow[8], lrow[8];
  v8f o[4];
  for (int j = 0; j < 8; j++) { mrow[j] = -1e30f; lrow[j] = 0.f; }
  for (int t = 0; t < 4; t++)
    for (int j = 0; j < 8; j++) o[t][j] = 0.f;

  __shared__ __align__(16) _Float16 Plds[16 * 32];

  for (int kc = 0; kc < NKV / 32; kc++) {
    if (kc + 1 < NKV / 32) {  // prefetch next chunk's K and V rows
      __builtin_prefetch(k16 + base + (size_t)((kc + 1) * 32 + lane) * HDIM, 0, 0);
      __builtin_prefetch(v16 + base + (size_t)((kc + 1) * 32 + lane) * HDIM, 0, 0);
    }
    v8f sc[2];
    for (int ct = 0; ct < 2; ct++) {
      Frag16 kf0, kf1;
      const _Float16* kp = k16 + base + (size_t)(kc * 32 + ct * 16 + l16) * HDIM;
      for (int jj = 0; jj < 8; jj++) {
        kf0.u[jj] = *(const unsigned*)&kp[kpair(jj, hi)];
        kf1.u[jj] = *(const unsigned*)&kp[kpair(jj, hi) + 32];
      }
      v8f z;
      for (int j = 0; j < 8; j++) z[j] = 0.f;
      z = __builtin_amdgcn_wmma_f32_16x16x32_f16(false, qa[0].h, false, kf0.h, (short)0, z, false, false);
      z = __builtin_amdgcn_wmma_f32_16x16x32_f16(false, qa[1].h, false, kf1.h, (short)0, z, false, false);
      int col = kc * 32 + ct * 16 + l16;
      for (int j = 0; j < 8; j++)
        sc[ct][j] = (col < NTOK) ? z[j] * 0.125f : -1e30f;
    }
    float p0[8], p1[8];
    for (int j = 0; j < 8; j++) {
      float cm = fmaxf(sc[0][j], sc[1][j]);
      for (int d = 1; d < 16; d <<= 1) cm = fmaxf(cm, __shfl_xor(cm, d, 16));
      float mn = fmaxf(mrow[j], cm);
      float alpha = expf(mrow[j] - mn);
      float a0 = expf(sc[0][j] - mn), a1 = expf(sc[1][j] - mn);
      float ssum = a0 + a1;
      for (int d = 1; d < 16; d <<= 1) ssum += __shfl_xor(ssum, d, 16);
      lrow[j] = lrow[j] * alpha + ssum;
      mrow[j] = mn;
      for (int t = 0; t < 4; t++) o[t][j] *= alpha;
      p0[j] = a0; p1[j] = a1;
    }
    __syncthreads();
    for (int j = 0; j < 8; j++) {
      int row = j + hi * 8;
      Plds[row * 32 + l16]      = (_Float16)p0[j];
      Plds[row * 32 + 16 + l16] = (_Float16)p1[j];
    }
    __syncthreads();
    Frag16 pf;
    {
      const _Float16* pp = &Plds[l16 * 32];
      for (int jj = 0; jj < 8; jj++) pf.u[jj] = *(const unsigned*)&pp[kpair(jj, hi)];
    }
    for (int t = 0; t < 4; t++) {
      Frag16 vf;
      for (int j = 0; j < 16; j++) {
        int kv = kc * 32 + kone(j, hi);
        vf.e[j] = v16[base + (size_t)kv * HDIM + t * 16 + l16];
      }
      o[t] = __builtin_amdgcn_wmma_f32_16x16x32_f16(false, pf.h, false, vf.h, (short)0, o[t], false, false);
    }
  }

  int s = sh >> 4, h = sh & 15;
  for (int t = 0; t < 4; t++)
    for (int j = 0; j < 8; j++) {
      int row = qt * 16 + j + hi * 8;
      if (row < NTOK) {
        size_t rr = (size_t)(s * NTOK + row) * 4096 + h * HDIM + t * 16 + l16;
        outA[rr] = (_Float16)(o[t][j] / lrow[j]);
      }
    }
}

// ---------------- residual with layer-scale: X += ls * Y ----------------
__global__ void resid_kernel(float* __restrict__ X, const float* __restrict__ Y,
                             const float* __restrict__ ls) {
  int idx = blockIdx.x * blockDim.x + threadIdx.x;
  if (idx >= MROWS * EMB) return;
  int c = idx & 1023;
  X[idx] += ls[c] * Y[idx];
}

// ---------------- exact GeLU + convert to f16 (lda=4096 both sides) ----------
__global__ void gelu_f16_kernel(const float* __restrict__ in, _Float16* __restrict__ out) {
  int idx = blockIdx.x * blockDim.x + threadIdx.x;
  if (idx >= MROWS * 4096) return;
  float x = in[idx];
  out[idx] = (_Float16)(0.5f * x * (1.f + erff(x * 0.70710678118654752f)));
}

// ---------------- final gather to (L,S,P,C) ----------------
__global__ void output_kernel(const float* __restrict__ X, float* __restrict__ out) {
  int idx = blockIdx.x * blockDim.x + threadIdx.x;
  if (idx >= 2 * 2 * 729 * EMB) return;
  int c = idx & 1023;
  int p = (idx >> 10) % 729;
  int s = (idx >> 10) / 729 % 2;
  int l = (idx >> 10) / (729 * 2);
  out[idx] = X[((size_t)(s * NTOK + l * 733 + 4 + p)) * EMB + c];
}

// =======================================================================
extern "C" void kernel_launch(void* const* d_in, const int* in_sizes, int n_in,
                              void* d_out, int out_size, void* d_ws, size_t ws_size,
                              hipStream_t stream) {
  (void)in_sizes; (void)n_in; (void)out_size; (void)ws_size;
  const float* pt  = (const float*)d_in[0];
  const float* reg = (const float*)d_in[1];
  // block params: d_in[2 + b*18 + k], dict order:
  // 0 norm1_w 1 norm1_b 2 qkv_w 3 qkv_b 4 q_norm_w 5 q_norm_b 6 k_norm_w 7 k_norm_b
  // 8 proj_w 9 proj_b 10 ls1 11 norm2_w 12 norm2_b 13 fc1_w 14 fc1_b 15 fc2_w 16 fc2_b 17 ls2

  // workspace carve
  char* w = (char*)d_ws;
  auto carve = [&](size_t bytes) -> void* {
    void* p = (void*)w;
    w += (bytes + 255) & ~(size_t)255;
    return p;
  };
  float*     X    = (float*)    carve((size_t)MROWS * EMB * 4);
  _Float16*  A16  = (_Float16*) carve((size_t)MPAD * 4096 * 2);
  float*     big  = (float*)    carve((size_t)MPAD * 4096 * 4);
  float*     Yb   = (float*)    carve((size_t)MPAD * EMB * 4);
  size_t qkvsz = (size_t)32 * NKV * HDIM;        // elements
  _Float16*  q16  = (_Float16*) carve(qkvsz * 2);
  _Float16*  k16  = (_Float16*) carve(qkvsz * 2);
  _Float16*  v16  = (_Float16*) carve(qkvsz * 2);
  _Float16*  Wt[2][4];
  const int  wK[4] = {1024, 1024, 1024, 4096};
  const int  wN[4] = {3072, 1024, 4096, 1024};
  const int  wIdx[4] = {2, 8, 13, 15};           // qkv_w, proj_w, fc1_w, fc2_w
  for (int b = 0; b < 2; b++)
    for (int i = 0; i < 4; i++)
      Wt[b][i] = (_Float16*)carve((size_t)wK[i] * wN[i] * 2);

  // zero padding rows of A16 (rows MROWS..MPAD-1) and q/k/v pad rows
  hipMemsetAsync(A16 + (size_t)MROWS * 4096, 0, (size_t)(MPAD - MROWS) * 4096 * 2, stream);
  hipMemsetAsync(q16, 0, qkvsz * 2, stream);
  hipMemsetAsync(k16, 0, qkvsz * 2, stream);
  hipMemsetAsync(v16, 0, qkvsz * 2, stream);

  // tokens
  {
    int total = MROWS * EMB;
    build_tokens_kernel<<<(total + 255) / 256, 256, 0, stream>>>(pt, reg, X);
  }
  // weights -> f16 transposed
  for (int b = 0; b < 2; b++)
    for (int i = 0; i < 4; i++) {
      const float* src = (const float*)d_in[2 + b * 18 + wIdx[i]];
      int total = wK[i] * wN[i];
      convt_kernel<<<(total + 255) / 256, 256, 0, stream>>>(src, Wt[b][i], wK[i], wN[i]);
    }

  dim3 blk256(256);
  for (int b = 0; b < 2; b++) {
    const float* const* P = (const float* const*)(d_in + 2 + b * 18);
    // ---- attention half ----
    layernorm_f16_kernel<<<MROWS, 256, 0, stream>>>(X, P[0], P[1], A16);
    gemm_f16_wmma_kernel<<<dim3(3072 / 128, MPAD / 128), blk256, 0, stream>>>(
        A16, Wt[b][0], big, P[3], 3072, 1024);
    qkv_pack_rope_kernel<<<dim3(MROWS, NHEAD), 32, 0, stream>>>(
        big, P[4], P[5], P[6], P[7], q16, k16, v16);
    flash_attn_kernel<<<dim3(NKV / 16, 32), 32, 0, stream>>>(q16, k16, v16, A16);
    gemm_f16_wmma_kernel<<<dim3(1024 / 128, MPAD / 128), blk256, 0, stream>>>(
        A16, Wt[b][1], Yb, P[9], 1024, 1024);
    resid_kernel<<<(MROWS * EMB + 255) / 256, 256, 0, stream>>>(X, Yb, P[10]);
    // ---- MLP half ----
    layernorm_f16_kernel<<<MROWS, 256, 0, stream>>>(X, P[11], P[12], A16);
    gemm_f16_wmma_kernel<<<dim3(4096 / 128, MPAD / 128), blk256, 0, stream>>>(
        A16, Wt[b][2], big, P[14], 4096, 1024);
    gelu_f16_kernel<<<(MROWS * 4096 + 255) / 256, 256, 0, stream>>>(big, A16);
    gemm_f16_wmma_kernel<<<dim3(1024 / 128, MPAD / 128), blk256, 0, stream>>>(
        A16, Wt[b][3], Yb, P[16], 1024, 4096);
    resid_kernel<<<(MROWS * EMB + 255) / 256, 256, 0, stream>>>(X, Yb, P[17]);
  }
  output_kernel<<<(2 * 2 * 729 * EMB + 255) / 256, 256, 0, stream>>>(X, (float*)d_out);
}
